// BehaviorAttention_19353122636370
// MI455X (gfx1250) — matrix operand order repaired and checked by reading, not compiled, not run on hardware
//
#include <hip/hip_runtime.h>

#define S_LEN 2048
#define NB 8
#define DIM 1024
#define SCALE_F 0.03125f  // 1/sqrt(1024)

typedef __attribute__((ext_vector_type(16))) __bf16       v16bf;
typedef __attribute__((ext_vector_type(8)))  float        v8f;
typedef __attribute__((ext_vector_type(4)))  unsigned int u32x4;
typedef __attribute__((ext_vector_type(4)))  float        f32x4;

union Frag {
  v16bf bf;
  struct { u32x4 lo; u32x4 hi; } q;
};

static __device__ inline v8f v8f_zero() {
  v8f z = {0.f, 0.f, 0.f, 0.f, 0.f, 0.f, 0.f, 0.f};
  return z;
}

static __device__ inline unsigned short bf16_bits(float x) {
  union { __bf16 h; unsigned short u; } cv;
  cv.h = (__bf16)x;
  return cv.u;
}

// ---------------------------------------------------------------------------
// fp32 -> bf16 weight conversion
// ---------------------------------------------------------------------------
__global__ __launch_bounds__(256) void f32_to_bf16_kernel(
    const float* __restrict__ src, unsigned short* __restrict__ dst, int n) {
  int i = blockIdx.x * 256 + threadIdx.x;
  if (i < n) dst[i] = bf16_bits(src[i]);
}

// ---------------------------------------------------------------------------
// Projection GEMM: Out = X @ W^T + bias   (X fp32 [16384, D], W bf16 [D, D])
// Each wave computes a 16x64 tile with v_wmma_f32_16x16x32_bf16.
// TRANS_OUT=false: out bf16 [n, m] row-major (n = s*NB + batch).
// TRANS_OUT=true : out bf16 written as Vt[batch][m][s]  (for the PV GEMM).
// ---------------------------------------------------------------------------
template <bool TRANS_OUT>
__global__ __launch_bounds__(128) void proj_kernel(
    const float* __restrict__ X, const unsigned short* __restrict__ Wb,
    const float* __restrict__ bias, unsigned short* __restrict__ Out) {
  const int tid = threadIdx.x;
  const int wave = tid >> 5, lane = tid & 31;
  const int lhalf = lane >> 4, lmod = lane & 15;
  const int n0 = blockIdx.x * 16;
  const int m0 = blockIdx.y * 256 + wave * 64;

  v8f acc[4];
  for (int t = 0; t < 4; ++t) acc[t] = v8f_zero();

  const int arow = n0 + lmod;
  const int akb = lhalf * 8;    // A frag: K halves at akb+0..7 and akb+16..23
  const int bkb = lhalf * 8;    // B frag: K halves at bkb+0..7 and bkb+16..23

  for (int k0 = 0; k0 < DIM; k0 += 32) {
    // A fragment: load fp32, convert to bf16 in registers
    const float* xp = &X[(size_t)arow * DIM + k0 + akb];
    f32x4 x0 = *(const f32x4*)xp;
    f32x4 x1 = *(const f32x4*)(xp + 4);
    f32x4 x2 = *(const f32x4*)(xp + 16);
    f32x4 x3 = *(const f32x4*)(xp + 20);
    Frag a;
    for (int e = 0; e < 4; ++e) {
      a.bf[e]      = (__bf16)x0[e];
      a.bf[4 + e]  = (__bf16)x1[e];
      a.bf[8 + e]  = (__bf16)x2[e];
      a.bf[12 + e] = (__bf16)x3[e];
    }
    for (int nt = 0; nt < 4; ++nt) {
      const unsigned short* wp =
          &Wb[(size_t)(m0 + nt * 16 + lmod) * DIM + k0 + bkb];
      Frag b;
      b.q.lo = *(const u32x4*)wp;
      b.q.hi = *(const u32x4*)(wp + 16);
      acc[nt] = __builtin_amdgcn_wmma_f32_16x16x32_bf16(
          false, a.bf, false, b.bf, (short)0, acc[nt], false, false);
    }
  }

  for (int nt = 0; nt < 4; ++nt) {
    const int m = m0 + nt * 16 + lmod;
    const float bval = bias[m];
    for (int e = 0; e < 8; ++e) {
      const float v = acc[nt][e] + bval;
      const int n = n0 + e + lhalf * 8;
      if (TRANS_OUT) {
        const int srow = n >> 3, bat = n & 7;  // n = s*NB + batch, NB==8
        Out[((size_t)bat * DIM + m) * S_LEN + srow] = bf16_bits(v);
      } else {
        Out[(size_t)n * DIM + m] = bf16_bits(v);
      }
    }
  }
}

// ---------------------------------------------------------------------------
// Output projection: d_out(fp32) = Ctx(bf16) @ Wo^T + bo
// ---------------------------------------------------------------------------
__global__ __launch_bounds__(128) void oproj_kernel(
    const unsigned short* __restrict__ Ctx, const unsigned short* __restrict__ Wb,
    const float* __restrict__ bias, float* __restrict__ Out) {
  const int tid = threadIdx.x;
  const int wave = tid >> 5, lane = tid & 31;
  const int lhalf = lane >> 4, lmod = lane & 15;
  const int n0 = blockIdx.x * 16;
  const int m0 = blockIdx.y * 256 + wave * 64;

  v8f acc[4];
  for (int t = 0; t < 4; ++t) acc[t] = v8f_zero();

  const int arow = n0 + lmod;
  const int akb = lhalf * 8;
  const int bkb = lhalf * 8;

  for (int k0 = 0; k0 < DIM; k0 += 32) {
    const unsigned short* cp = &Ctx[(size_t)arow * DIM + k0 + akb];
    Frag a;
    a.q.lo = *(const u32x4*)cp;
    a.q.hi = *(const u32x4*)(cp + 16);
    for (int nt = 0; nt < 4; ++nt) {
      const unsigned short* wp =
          &Wb[(size_t)(m0 + nt * 16 + lmod) * DIM + k0 + bkb];
      Frag b;
      b.q.lo = *(const u32x4*)wp;
      b.q.hi = *(const u32x4*)(wp + 16);
      acc[nt] = __builtin_amdgcn_wmma_f32_16x16x32_bf16(
          false, a.bf, false, b.bf, (short)0, acc[nt], false, false);
    }
  }

  for (int nt = 0; nt < 4; ++nt) {
    const int m = m0 + nt * 16 + lmod;
    const float bval = bias[m];
    for (int e = 0; e < 8; ++e) {
      const int n = n0 + e + lhalf * 8;
      Out[(size_t)n * DIM + m] = acc[nt][e] + bval;
    }
  }
}

// ---------------------------------------------------------------------------
// Flash attention: per (batch, 16-query block), stream key blocks of 32 with
// online softmax. 4 waves: each owns a 256-wide D-slice of the context.
// Q: bf16 [S,NB,D]; K: bf16 [S,NB,D]; Vt: bf16 [NB,D,S]; Ctx out: bf16 [S,NB,D]
// ---------------------------------------------------------------------------
__global__ __launch_bounds__(128) void flash_kernel(
    const unsigned short* __restrict__ Qb, const unsigned short* __restrict__ Kb,
    const unsigned short* __restrict__ Vt, unsigned short* __restrict__ Ctx) {
  __shared__ __align__(16) unsigned short qtile[16 * DIM];   // 32 KB
  __shared__ __align__(16) float spart[4][16][32];           // 8 KB
  __shared__ __align__(16) unsigned short ptile[16 * 32];    // 1 KB

  const int i0 = blockIdx.x * 16;
  const int batch = blockIdx.y;
  const int tid = threadIdx.x;
  const int wave = tid >> 5, lane = tid & 31;
  const int lhalf = lane >> 4, lmod = lane & 15;

  // Stage the whole 16 x 1024 Q tile into LDS (b128 chunks).
  for (int chunk = tid; chunk < 16 * DIM / 8; chunk += 128) {
    const int r = chunk >> 7;            // DIM/8 == 128 chunks per row
    const int dcol = (chunk & 127) * 8;
    *(u32x4*)&qtile[r * DIM + dcol] =
        *(const u32x4*)&Qb[((size_t)(i0 + r) * NB + batch) * DIM + dcol];
  }
  __syncthreads();

  v8f acc[16];
  for (int t = 0; t < 16; ++t) acc[t] = v8f_zero();
  float mrow[8], lrow[8];
  for (int r = 0; r < 8; ++r) { mrow[r] = -1e30f; lrow[r] = 0.0f; }

  const size_t vbase = (size_t)batch * DIM * S_LEN;
  const int akb = lhalf * 8;
  const int bkb = lhalf * 8;

  for (int j0 = 0; j0 < i0 + 16; j0 += 32) {
    // ---- stage 1: partial scores over this wave's 256-wide D slice ----
    v8f c0 = v8f_zero(), c1 = v8f_zero();
    for (int kk = 0; kk < 8; ++kk) {
      const int d0 = wave * 256 + kk * 32;
      Frag a;
      a.q.lo = *(const u32x4*)&qtile[lmod * DIM + d0 + akb];
      a.q.hi = *(const u32x4*)&qtile[lmod * DIM + d0 + akb + 16];
      Frag b0, b1;
      {
        const unsigned short* kp =
            &Kb[((size_t)(j0 + lmod) * NB + batch) * DIM + d0 + bkb];
        b0.q.lo = *(const u32x4*)kp;
        b0.q.hi = *(const u32x4*)(kp + 16);
      }
      {
        const unsigned short* kp =
            &Kb[((size_t)(j0 + 16 + lmod) * NB + batch) * DIM + d0 + bkb];
        b1.q.lo = *(const u32x4*)kp;
        b1.q.hi = *(const u32x4*)(kp + 16);
      }
      c0 = __builtin_amdgcn_wmma_f32_16x16x32_bf16(false, a.bf, false, b0.bf,
                                                   (short)0, c0, false, false);
      c1 = __builtin_amdgcn_wmma_f32_16x16x32_bf16(false, a.bf, false, b1.bf,
                                                   (short)0, c1, false, false);
    }
    // publish partials
    float* sp = &spart[wave][0][0];
    for (int t = 0; t < 8; ++t) {
      sp[t * 32 + lane] = c0[t];
      sp[(t + 8) * 32 + lane] = c1[t];
    }
    __syncthreads();

    // ---- cross-wave reduce + online softmax (redundant per wave) ----
    float s[16];
    for (int t = 0; t < 16; ++t) {
      float acc_s = spart[0][t][lane] + spart[1][t][lane] +
                    spart[2][t][lane] + spart[3][t][lane];
      s[t] = acc_s * SCALE_F;
    }
    if (j0 + 31 > i0) {  // causal mask only near the diagonal
      for (int t = 0; t < 16; ++t) {
        const int row = i0 + (t & 7) + lhalf * 8;
        const int col = j0 + ((t >> 3) * 16) + lmod;
        if (col > row) s[t] = -1e30f;
      }
    }
    float scfac[8];
    for (int r = 0; r < 8; ++r) {
      float v = fmaxf(s[r], s[r + 8]);
      v = fmaxf(v, __shfl_xor(v, 1, 32));
      v = fmaxf(v, __shfl_xor(v, 2, 32));
      v = fmaxf(v, __shfl_xor(v, 4, 32));
      v = fmaxf(v, __shfl_xor(v, 8, 32));
      const float mn = fmaxf(mrow[r], v);
      scfac[r] = __expf(mrow[r] - mn);
      mrow[r] = mn;
    }
    float p[16];
    for (int t = 0; t < 16; ++t) p[t] = __expf(s[t] - mrow[t & 7]);
    for (int r = 0; r < 8; ++r) {
      float v = p[r] + p[r + 8];
      v += __shfl_xor(v, 1, 32);
      v += __shfl_xor(v, 2, 32);
      v += __shfl_xor(v, 4, 32);
      v += __shfl_xor(v, 8, 32);
      lrow[r] = lrow[r] * scfac[r] + v;
    }
    if (wave == 0) {  // one wave publishes P (bf16, row-major 16x32)
      for (int t = 0; t < 16; ++t) {
        const int row = (t & 7) + lhalf * 8;
        const int col = ((t >> 3) * 16) + lmod;
        ptile[row * 32 + col] = bf16_bits(p[t]);
      }
    }
    __syncthreads();

    // ---- stage 2: rescale accumulators, Ctx_slice += P @ V_slice ----
    for (int t = 0; t < 16; ++t)
      for (int e = 0; e < 8; ++e) acc[t][e] *= scfac[e];
    Frag pa;
    pa.q.lo = *(const u32x4*)&ptile[lmod * 32 + akb];
    pa.q.hi = *(const u32x4*)&ptile[lmod * 32 + akb + 16];
    for (int nt = 0; nt < 16; ++nt) {
      const int d = wave * 256 + nt * 16 + lmod;
      const unsigned short* vp = &Vt[vbase + (size_t)d * S_LEN + j0 + bkb];
      Frag vb;
      vb.q.lo = *(const u32x4*)vp;
      vb.q.hi = *(const u32x4*)(vp + 16);
      acc[nt] = __builtin_amdgcn_wmma_f32_16x16x32_bf16(
          false, pa.bf, false, vb.bf, (short)0, acc[nt], false, false);
    }
    __syncthreads();  // protect spart/ptile for next key block
  }

  // finalize: divide by row sums, write bf16 context slice
  float linv[8];
  for (int r = 0; r < 8; ++r) linv[r] = 1.0f / lrow[r];
  for (int nt = 0; nt < 16; ++nt) {
    const int gcol = wave * 256 + nt * 16 + lmod;
    for (int e = 0; e < 8; ++e) {
      const int grow = i0 + e + lhalf * 8;
      Ctx[((size_t)grow * NB + batch) * DIM + gcol] =
          bf16_bits(acc[nt][e] * linv[e]);
    }
  }
}

// ---------------------------------------------------------------------------
extern "C" void kernel_launch(void* const* d_in, const int* in_sizes, int n_in,
                              void* d_out, int out_size, void* d_ws,
                              size_t ws_size, hipStream_t stream) {
  (void)in_sizes; (void)n_in; (void)out_size; (void)ws_size;
  const float* query = (const float*)d_in[0];
  const float* key_i = (const float*)d_in[1];
  const float* value = (const float*)d_in[2];
  // d_in[3] = causal mask (implicit in the kernel)
  const float* Wq = (const float*)d_in[4];
  const float* bq = (const float*)d_in[5];
  const float* Wk = (const float*)d_in[6];
  const float* bk = (const float*)d_in[7];
  const float* Wv = (const float*)d_in[8];
  const float* bv = (const float*)d_in[9];
  const float* Wo = (const float*)d_in[10];
  const float* bo = (const float*)d_in[11];

  unsigned short* ws = (unsigned short*)d_ws;
  const size_t WN = (size_t)DIM * DIM;          // 1M elems per weight
  const size_t TN = (size_t)S_LEN * NB * DIM;   // 16M elems per tensor
  unsigned short* WqB = ws;
  unsigned short* WkB = WqB + WN;
  unsigned short* WvB = WkB + WN;
  unsigned short* WoB = WvB + WN;
  unsigned short* Qb = WoB + WN;
  unsigned short* Kb = Qb + TN;
  unsigned short* Vt = Kb + TN;
  unsigned short* Cx = Vt + TN;   // total ws usage: 8 MB + 128 MB

  dim3 cg((unsigned)((WN + 255) / 256));
  f32_to_bf16_kernel<<<cg, 256, 0, stream>>>(Wq, WqB, (int)WN);
  f32_to_bf16_kernel<<<cg, 256, 0, stream>>>(Wk, WkB, (int)WN);
  f32_to_bf16_kernel<<<cg, 256, 0, stream>>>(Wv, WvB, (int)WN);
  f32_to_bf16_kernel<<<cg, 256, 0, stream>>>(Wo, WoB, (int)WN);

  dim3 pg(S_LEN * NB / 16, DIM / 256);  // (1024, 4), 4 waves/block
  proj_kernel<false><<<pg, 128, 0, stream>>>(query, WqB, bq, Qb);
  proj_kernel<false><<<pg, 128, 0, stream>>>(key_i, WkB, bk, Kb);
  proj_kernel<true><<<pg, 128, 0, stream>>>(value, WvB, bv, Vt);

  dim3 fg(S_LEN / 16, NB);              // (128, 8)
  flash_kernel<<<fg, 128, 0, stream>>>(Qb, Kb, Vt, Cx);

  oproj_kernel<<<pg, 128, 0, stream>>>(Cx, WoB, bo, (float*)d_out);
}
